// Decoder_79310866088114
// MI455X (gfx1250) — matrix operand order repaired and checked by reading, not compiled, not run on hardware
//
#include <hip/hip_runtime.h>
#include <hip/hip_bf16.h>

// ---------------------------------------------------------------------------
// GRU decoder for MI455X (gfx1250, wave32).
//   - W_hh converted once to bf16 in a WMMA A-fragment-swizzled layout
//     (100.7 MB -> resident in the 192 MB L2 across all 2048 steps).
//   - Persistent kernel, atomic grid barrier, 2 barriers per step.
//   - Per-step GEMV done with v_wmma_f32_16x16x32_bf16 (h broadcast across N).
// ---------------------------------------------------------------------------

typedef __attribute__((ext_vector_type(16))) __bf16 v16bf;
typedef __attribute__((ext_vector_type(8)))  float  v8f;

#define NWG          48
#define TPB          256
#define WAVES_PER_WG 8
#define NWAVES       (NWG * WAVES_PER_WG)   // 384 waves
#define HDIM         4096
#define G3           12288                  // 3*H rows of W_hh
#define KTILES       (HDIM / 32)            // 128 k-tiles of 32
#define RTILES       (G3 / 16)              // 768 row-tiles of 16
#define RT_PER_WAVE  (RTILES / NWAVES)      // 2

__device__ __forceinline__ unsigned short f2bf(float f) {
    unsigned u = __float_as_uint(f);
    u = u + 0x7FFFu + ((u >> 16) & 1u);      // round to nearest even
    return (unsigned short)(u >> 16);
}

__device__ __forceinline__ float sigmoidf_(float x) {
    return 1.0f / (1.0f + __expf(-x));
}

__device__ __forceinline__ void grid_barrier(unsigned* bar, unsigned target) {
    __threadfence();                         // publish prior global writes (device scope)
    __syncthreads();
    if (threadIdx.x == 0) {
        __hip_atomic_fetch_add(bar, 1u, __ATOMIC_ACQ_REL, __HIP_MEMORY_SCOPE_AGENT);
        while (__hip_atomic_load(bar, __ATOMIC_ACQUIRE, __HIP_MEMORY_SCOPE_AGENT) < target)
            __builtin_amdgcn_s_sleep(2);
    }
    __syncthreads();
    __threadfence();                         // acquire side for all threads
}

// ---- prep: h0 = encoder_hidden --------------------------------------------
__global__ void init_h_kernel(const float* __restrict__ enc, float* __restrict__ h) {
    int i = blockIdx.x * blockDim.x + threadIdx.x;
    if (i < HDIM) h[i] = enc[i];
}

// ---- prep: gx = W_ih @ relu(x) + b_ih  (12288 x 400, one row per thread) --
__global__ __launch_bounds__(TPB) void gx_kernel(const float* __restrict__ x400,
                                                 const float* __restrict__ W_ih,
                                                 const float* __restrict__ b_ih,
                                                 float* __restrict__ gx) {
    __shared__ float xs[400];
    for (int j = threadIdx.x; j < 400; j += TPB) {
        float v = x400[j];
        xs[j] = v > 0.0f ? v : 0.0f;
    }
    __syncthreads();
    int r = blockIdx.x * TPB + threadIdx.x;
    if (r < G3) {
        const float* w = W_ih + (size_t)r * 400;
        float acc = 0.0f;
        for (int j = 0; j < 400; ++j) acc += w[j] * xs[j];
        gx[r] = acc + b_ih[r];
    }
}

// ---- prep: swizzle W_hh (fp32 row-major) into bf16 WMMA A-fragment tiles --
// Tile (rt,kt) occupies 1024 B; lane owns 32 B = 16 bf16 slots.
// 16-bit A layout (CDNA5 ISA): lanes 0-15: M=lane, v0..3 -> K=2v+h,
// v4..7 -> K=16+2(v-4)+h ; lanes 16-31: M=lane-16, +8 / +24.
__global__ __launch_bounds__(TPB) void swizzle_whh_kernel(const float* __restrict__ W_hh,
                                                          unsigned short* __restrict__ wt) {
    unsigned id   = blockIdx.x * TPB + threadIdx.x;   // tile*32 + lane
    unsigned tile = id >> 5;
    unsigned lane = id & 31u;
    if (tile >= RTILES * KTILES) return;
    unsigned rt  = tile >> 7;          // / KTILES
    unsigned kt  = tile & 127u;        // % KTILES
    unsigned m   = rt * 16 + (lane & 15u);
    unsigned grp = lane >> 4;
    const float* src = W_hh + (size_t)m * HDIM + (size_t)kt * 32;
    unsigned short* dst = wt + (size_t)tile * 512 + (size_t)lane * 16;
#pragma unroll
    for (int j = 0; j < 16; ++j) {
        unsigned v  = (unsigned)j >> 1;
        unsigned hh = (unsigned)j & 1u;
        unsigned k_local = ((v & 4u) ? 16u : 0u) + grp * 8u + ((v & 3u) << 1) + hh;
        dst[j] = f2bf(src[k_local]);
    }
}

// ---- persistent GRU decode -------------------------------------------------
__global__ __launch_bounds__(TPB) void gru_decode_kernel(
    const unsigned short* __restrict__ wt,    // swizzled bf16 W_hh (L2 resident)
    const float* __restrict__ gx,             // 3H, includes b_ih
    const float* __restrict__ b_hh,           // 3H
    const float* __restrict__ W_out,          // 401 x 4096
    const float* __restrict__ b_out,          // 401
    float* __restrict__ h,                    // 4096, updated in place
    float* __restrict__ gh,                   // 3H scratch
    unsigned* __restrict__ bar,
    const int* __restrict__ maxlen,
    float* __restrict__ out)                  // [T*400 outputs][T stops]
{
    __shared__ __align__(128) unsigned short h_lds[HDIM];   // h in bf16
    const int tid  = threadIdx.x;
    const int lane = tid & 31;
    const int wave = blockIdx.x * WAVES_PER_WG + (tid >> 5);
    const int T    = *maxlen;
    unsigned target = 0;

    for (int t = 0; t < T; ++t) {
        // ---- stage h into LDS as bf16 (B operand source) ------------------
        {
            const int per = HDIM / TPB;      // 16
            int base = tid * per;
#pragma unroll
            for (int j = 0; j < per; ++j)
                h_lds[base + j] = f2bf(h[base + j]);
        }
        __syncthreads();

        // ---- phase 1: gh = W_hh @ h via WMMA ------------------------------
#pragma unroll
        for (int i = 0; i < RT_PER_WAVE; ++i) {
            int rt = wave * RT_PER_WAVE + i;
            const unsigned short* wrow = wt + (size_t)rt * KTILES * 512 + (size_t)lane * 16;
            const unsigned short* bsrc = h_lds + ((lane >> 4) << 3);
            v8f c = {};
#pragma unroll 4
            for (int kt = 0; kt < KTILES; ++kt) {
                v16bf a = *(const v16bf*)(wrow);
                typedef __attribute__((ext_vector_type(8))) __bf16 v8bf_;
                v8bf_ b_lo = *(const v8bf_*)(bsrc);
                v8bf_ b_hi = *(const v8bf_*)(bsrc + 16);
                v16bf b = __builtin_shufflevector(b_lo, b_hi, 0, 1, 2, 3, 4, 5, 6, 7, 8, 9, 10, 11, 12, 13, 14, 15);
                __builtin_prefetch(wrow + 4 * 512, 0, 0);   // global_prefetch_b8
                c = __builtin_amdgcn_wmma_f32_16x16x32_bf16(
                        false, a, false, b, (short)0, c, false, false);
                wrow += 512;
                bsrc += 32;
            }
            // D cols all equal (h replicated across N): lane 0 -> rows 0..7,
            // lane 16 -> rows 8..15 of this row-tile.
            if ((lane & 15) == 0) {
                float* dst = gh + rt * 16 + ((lane >> 4) << 3);
                dst[0] = c[0]; dst[1] = c[1]; dst[2] = c[2]; dst[3] = c[3];
                dst[4] = c[4]; dst[5] = c[5]; dst[6] = c[6]; dst[7] = c[7];
            }
        }
        grid_barrier(bar, target += NWG);

        // ---- phase 2: gates + h update (first 4096 threads) ---------------
        {
            int i = blockIdx.x * TPB + tid;
            if (i < HDIM) {
                float ghr = gh[i]            + b_hh[i];
                float ghz = gh[HDIM + i]     + b_hh[HDIM + i];
                float ghn = gh[2 * HDIM + i] + b_hh[2 * HDIM + i];
                float r = sigmoidf_(gx[i] + ghr);
                float z = sigmoidf_(gx[HDIM + i] + ghz);
                float n = tanhf(gx[2 * HDIM + i] + r * ghn);
                h[i] = (1.0f - z) * n + z * h[i];
            }
        }
        grid_barrier(bar, target += NWG);

        // ---- phase 3: o = W_out @ h_new + b_out; tanh / sigmoid -----------
        for (int row = wave; row < 401; row += NWAVES) {
            const float* wr = W_out + (size_t)row * HDIM;
            float acc = 0.0f;
            for (int k = lane; k < HDIM; k += 32)
                acc += wr[k] * h[k];
#pragma unroll
            for (int s = 16; s > 0; s >>= 1)
                acc += __shfl_xor(acc, s, 32);
            if (lane == 0) {
                float o = acc + b_out[row];
                if (row < 400) out[(size_t)t * 400 + row] = tanhf(o);
                else           out[(size_t)T * 400 + t]   = sigmoidf_(o);
            }
        }
        // next phase-1 barrier also orders phase-3 reads vs next h write
    }
}

// ---------------------------------------------------------------------------
extern "C" void kernel_launch(void* const* d_in, const int* in_sizes, int n_in,
                              void* d_out, int out_size, void* d_ws, size_t ws_size,
                              hipStream_t stream) {
    const float* start = (const float*)d_in[0];
    const float* enc   = (const float*)d_in[1];
    const float* W_ih  = (const float*)d_in[2];
    const float* W_hh  = (const float*)d_in[3];
    const float* b_ih  = (const float*)d_in[4];
    const float* b_hh  = (const float*)d_in[5];
    const float* W_out = (const float*)d_in[6];
    const float* b_out = (const float*)d_in[7];
    const int*   maxlen = (const int*)d_in[8];
    float* out = (float*)d_out;

    char* ws = (char*)d_ws;
    unsigned* bar = (unsigned*)ws;                              // 256 B
    float* h  = (float*)(ws + 256);                             // 4096 f32
    float* gx = h + HDIM;                                       // 12288 f32
    float* gh = gx + G3;                                        // 12288 f32
    unsigned short* wt = (unsigned short*)(ws + 256 + (size_t)(HDIM + 2 * G3) * 4);

    size_t need = 256 + (size_t)(HDIM + 2 * G3) * 4 + (size_t)G3 * HDIM * 2;
    if (ws_size < need) return;

    hipMemsetAsync(bar, 0, 256, stream);
    init_h_kernel<<<(HDIM + TPB - 1) / TPB, TPB, 0, stream>>>(enc, h);
    gx_kernel<<<NWG, TPB, 0, stream>>>(start, W_ih, b_ih, gx);
    swizzle_whh_kernel<<<(RTILES * KTILES * 32) / TPB, TPB, 0, stream>>>(W_hh, wt);
    gru_decode_kernel<<<NWG, TPB, 0, stream>>>(wt, gx, b_hh, W_out, b_out,
                                               h, gh, bar, maxlen, out);
}
